// WindowAttention_79929341378986
// MI455X (gfx1250) — matrix operand order repaired and checked by reading, not compiled, not run on hardware
//
#include <hip/hip_runtime.h>
#include <hip/hip_bf16.h>

typedef _Float16 half_t;
typedef __attribute__((ext_vector_type(8)))  _Float16 v8h;
typedef __attribute__((ext_vector_type(16))) _Float16 v16h;
typedef __attribute__((ext_vector_type(8)))  float    v8f;

// ---- workspace layout (bytes) ----------------------------------------------
#define CMB_OFF      0                    // f32 [64][6][49][49] mask+bias sums
#define QKVW_OFF     4194304              // f16 [576][192] qkv weights
#define PROJW_OFF    4415488              // f16 [192][192] proj weights
#define WS_QKV_BASE  8388608              // per-window q/k/v (f16)
#define WIN_HALVES   36864                // 3 * 6 heads * 64 * 32
#define HALVES_QK    2048                 // 64*32
// per window: q(h) at h*2048 [64][32], k(h) at (6+h)*2048 [64][32],
//             v(h) at (12+h)*2048 stored transposed [32][64].
// ctx [64][192] (n-major) overwrites the q region after attention.

// ---- WMMA operand loaders (contiguous k order, same in A and B) -------------------
// A (16x32 f16): lane r=lane&15 holds row r; elems i = K hi*16+i
//                (hi = lane>>4)
__device__ inline v16h load_a16(const half_t* base, int row, int stride_h,
                                int koff, int lane) {
  int r = lane & 15, hi = lane >> 4;
  const half_t* p = base + (size_t)(row + r) * stride_h + koff + hi * 16;
  v8h c0 = *(const v8h*)p;
  v8h c1 = *(const v8h*)(p + 8);
  return __builtin_shufflevector(c0, c1, 0,1,2,3,4,5,6,7,8,9,10,11,12,13,14,15);
}
// B (32x16 f16): lane holds column c=lane&15; elems i = K hi*16+i.
// Source matrix is [col][K] row-major (k tokens / v^T / weight rows).
__device__ inline v16h load_b16(const half_t* base, int row, int stride_h,
                                int koff, int lane) {
  int c = lane & 15, hi = lane >> 4;
  return *(const v16h*)(base + (size_t)(row + c) * stride_h + koff + hi * 16);
}

__device__ inline v8f wmma_f16(v16h a, v16h b, v8f c) {
  return __builtin_amdgcn_wmma_f32_16x16x32_f16(false, a, false, b,
                                                (short)0, c, false, false);
}

// ---- setup: weights f32 -> f16 (run once) ----------------------------------
__global__ void cvt_w_kernel(const float* __restrict__ qkv_w,
                             const float* __restrict__ proj_w,
                             half_t* __restrict__ qkv_wh,
                             half_t* __restrict__ proj_wh) {
  int idx = blockIdx.x * blockDim.x + threadIdx.x;
  if (idx < 576 * 192) qkv_wh[idx] = (half_t)qkv_w[idx];
  if (idx < 192 * 192) proj_wh[idx] = (half_t)proj_w[idx];
}

// ---- setup: combined (window mask + rel-pos bias) table [64][6][49][49] ----
__global__ void cmb_kernel(const float* __restrict__ mask,
                           const float* __restrict__ bias_table,
                           const int* __restrict__ rel_index,
                           float* __restrict__ cmb) {
  int idx = blockIdx.x * blockDim.x + threadIdx.x;
  if (idx < 64 * 6 * 2401) {
    int w = idx / (6 * 2401);
    int r = idx % (6 * 2401);
    int h = r / 2401, nm = r % 2401;
    cmb[idx] = mask[w * 2401 + nm] + bias_table[rel_index[nm] * 6 + h];
  }
}

// ---- kernel 1: QKV projection, one window per WG, 6 waves ------------------
__global__ __launch_bounds__(192) void qkv_kernel(
    const float* __restrict__ x, const half_t* __restrict__ qkv_wh,
    const float* __restrict__ qkv_b, half_t* __restrict__ ws_qkv) {
  __shared__ half_t xl[64 * 208];                 // x window in f16, padded
  const int win = blockIdx.x, tid = threadIdx.x;
  const int wave = tid >> 5, lane = tid & 31;
  const int c = lane & 15, hi = lane >> 4;

  const float* xw = x + (size_t)win * 49 * 192;
  for (int i = tid; i < 64 * 192; i += 192) {
    int r = i / 192, k = i % 192;
    xl[r * 208 + k] = (r < 49) ? (half_t)xw[i] : (half_t)0.f;
  }
  __syncthreads();

  half_t* wsw = ws_qkv + (size_t)win * WIN_HALVES;
  const float scale = 0.17677669529663687f;       // 32^-0.5

#pragma unroll 1   // keep one col-tile's working set live (< 256 VGPRs)
  for (int cti = 0; cti < 6; ++cti) {
    int ct = wave * 6 + cti;                      // output col-tile 0..35
    v16h bop[6];
#pragma unroll
    for (int kb = 0; kb < 6; ++kb)
      bop[kb] = load_b16(qkv_wh, ct * 16, 192, kb * 32, lane);
    float bias = qkv_b[ct * 16 + c];
    int three = ct / 12, rem = ct % 12;
    int h = rem >> 1, dbase = (rem & 1) * 16;
#pragma unroll 1   // limit live A-operand sets: one M-tile in flight
    for (int mt = 0; mt < 4; ++mt) {
      v8f acc;
#pragma unroll
      for (int j = 0; j < 8; ++j) acc[j] = bias;
#pragma unroll
      for (int kb = 0; kb < 6; ++kb)
        acc = wmma_f16(load_a16(xl, mt * 16, 208, kb * 32, lane), bop[kb], acc);
      if (three == 0) {
#pragma unroll
        for (int j = 0; j < 8; ++j) acc[j] *= scale;
      }
      if (three < 2) {                            // q,k: [64][32] token-major
        half_t* dst = wsw + (three * 6 + h) * HALVES_QK;
#pragma unroll
        for (int j = 0; j < 8; ++j)
          dst[(mt * 16 + hi * 8 + j) * 32 + dbase + c] = (half_t)acc[j];
      } else {                                    // v: transposed [32][64]
        half_t* dst = wsw + (12 + h) * HALVES_QK;
        v8h pk;
#pragma unroll
        for (int j = 0; j < 8; ++j) pk[j] = (half_t)acc[j];
        *(v8h*)(dst + (dbase + c) * 64 + mt * 16 + hi * 8) = pk;
      }
    }
  }
}

// ---- kernel 2: fused attention, one window per WG, wave h = head h ---------
__global__ __launch_bounds__(192) void attn_kernel(
    const float* __restrict__ cmb, half_t* __restrict__ ws_qkv) {
  __shared__ half_t pl[6][64 * 72];               // P (softmax probs) per wave
  const int win = blockIdx.x;
  const int h = threadIdx.x >> 5, lane = threadIdx.x & 31;
  const int c = lane & 15, hi = lane >> 4;

  half_t* wsw = ws_qkv + (size_t)win * WIN_HALVES;
  const half_t* qp = wsw + h * HALVES_QK;
  const half_t* kp = wsw + (6 + h) * HALVES_QK;
  const half_t* vp = wsw + (12 + h) * HALVES_QK;

  // S = q @ k^T   (64x64 padded, K=32)
  v16h aq[4], bk[4];
#pragma unroll
  for (int t = 0; t < 4; ++t) {
    aq[t] = load_a16(qp, t * 16, 32, 0, lane);
    bk[t] = load_b16(kp, t * 16, 32, 0, lane);
  }
  v8f S[4][4];
#pragma unroll
  for (int mt = 0; mt < 4; ++mt)
#pragma unroll
    for (int nt = 0; nt < 4; ++nt) {
      v8f z;
#pragma unroll
      for (int j = 0; j < 8; ++j) z[j] = 0.f;
      S[mt][nt] = wmma_f16(aq[mt], bk[nt], z);
    }

  // + combined (rel-pos bias + window mask); invalid columns -> -inf
  const float* cm = cmb + (size_t)((win & 63) * 6 + h) * 2401;
#pragma unroll
  for (int mt = 0; mt < 4; ++mt)
#pragma unroll
    for (int nt = 0; nt < 4; ++nt)
#pragma unroll
      for (int j = 0; j < 8; ++j) {
        int n = mt * 16 + hi * 8 + j, m = nt * 16 + c;
        float s = S[mt][nt][j];
        if (n < 49 && m < 49) s += cm[n * 49 + m];
        if (m >= 49) s = -1e30f;
        S[mt][nt][j] = s;
      }

  // row softmax: row n lives in one VGPR across a 16-lane half
#pragma unroll
  for (int mt = 0; mt < 4; ++mt)
#pragma unroll
    for (int j = 0; j < 8; ++j) {
      float mx = S[mt][0][j];
#pragma unroll
      for (int nt = 1; nt < 4; ++nt) mx = fmaxf(mx, S[mt][nt][j]);
#pragma unroll
      for (int off = 8; off >= 1; off >>= 1)
        mx = fmaxf(mx, __shfl_xor(mx, off, 32));
      float sum = 0.f;
#pragma unroll
      for (int nt = 0; nt < 4; ++nt) {
        float e = __expf(S[mt][nt][j] - mx);
        S[mt][nt][j] = e;
        sum += e;
      }
#pragma unroll
      for (int off = 8; off >= 1; off >>= 1) sum += __shfl_xor(sum, off, 32);
      float inv = 1.f / sum;
#pragma unroll
      for (int nt = 0; nt < 4; ++nt) S[mt][nt][j] *= inv;
    }

  // P -> LDS (f16) to re-shape into WMMA A-operand layout
  half_t* pw = pl[h];
#pragma unroll
  for (int mt = 0; mt < 4; ++mt)
#pragma unroll
    for (int nt = 0; nt < 4; ++nt)
#pragma unroll
      for (int j = 0; j < 8; ++j)
        pw[(mt * 16 + hi * 8 + j) * 72 + nt * 16 + c] = (half_t)S[mt][nt][j];
  asm volatile("s_wait_dscnt 0" ::: "memory");

  // O = P @ v    (K = 64, v stored transposed [32][64])
  v16h bv[2][2];
#pragma unroll
  for (int kb = 0; kb < 2; ++kb)
#pragma unroll
    for (int nt2 = 0; nt2 < 2; ++nt2)
      bv[kb][nt2] = load_b16(vp, nt2 * 16, 64, kb * 32, lane);
  v8f O[4][2];
#pragma unroll
  for (int mt = 0; mt < 4; ++mt) {
    v16h ap0 = load_a16(pw, mt * 16, 72, 0, lane);
    v16h ap1 = load_a16(pw, mt * 16, 72, 32, lane);
#pragma unroll
    for (int nt2 = 0; nt2 < 2; ++nt2) {
      v8f z;
#pragma unroll
      for (int j = 0; j < 8; ++j) z[j] = 0.f;
      z = wmma_f16(ap0, bv[0][nt2], z);
      O[mt][nt2] = wmma_f16(ap1, bv[1][nt2], z);
    }
  }

  __syncthreads();   // every wave done reading q before ctx overwrites it
  half_t* ctx = wsw; // ctx [n][192] overwrites the q region (same size)
#pragma unroll
  for (int mt = 0; mt < 4; ++mt)
#pragma unroll
    for (int nt2 = 0; nt2 < 2; ++nt2)
#pragma unroll
      for (int j = 0; j < 8; ++j) {
        int n = mt * 16 + hi * 8 + j;
        if (n < 49)
          ctx[n * 192 + h * 32 + nt2 * 16 + c] = (half_t)O[mt][nt2][j];
      }
}

// ---- kernel 3: output projection -------------------------------------------
__global__ __launch_bounds__(192) void proj_kernel(
    const half_t* __restrict__ proj_wh, const float* __restrict__ proj_b,
    const half_t* __restrict__ ws_qkv, float* __restrict__ out) {
  __shared__ half_t cl[64 * 208];
  const int win = blockIdx.x, tid = threadIdx.x;
  const int wave = tid >> 5, lane = tid & 31;
  const int c = lane & 15, hi = lane >> 4;

  const half_t* ctx = ws_qkv + (size_t)win * WIN_HALVES;
  for (int i = tid; i < 64 * 192; i += 192)
    cl[(i / 192) * 208 + (i % 192)] = ctx[i];   // pad rows hold stale finite data
  __syncthreads();

  float* ow = out + (size_t)win * 49 * 192;
#pragma unroll 1   // keep one col-tile's working set live (< 256 VGPRs)
  for (int nti = 0; nti < 2; ++nti) {
    int nt = wave * 2 + nti;                    // 0..11
    v16h bop[6];
#pragma unroll
    for (int kb = 0; kb < 6; ++kb)
      bop[kb] = load_b16(proj_wh, nt * 16, 192, kb * 32, lane);
    float bias = proj_b[nt * 16 + c];
#pragma unroll 1   // limit live A-operand sets: one M-tile in flight
    for (int mt = 0; mt < 4; ++mt) {
      v8f acc;
#pragma unroll
      for (int j = 0; j < 8; ++j) acc[j] = bias;
#pragma unroll
      for (int kb = 0; kb < 6; ++kb)
        acc = wmma_f16(load_a16(cl, mt * 16, 208, kb * 32, lane), bop[kb], acc);
#pragma unroll
      for (int j = 0; j < 8; ++j) {
        int n = mt * 16 + hi * 8 + j;
        if (n < 49) ow[n * 192 + nt * 16 + c] = acc[j];
      }
    }
  }
}

extern "C" void kernel_launch(void* const* d_in, const int* in_sizes, int n_in,
                              void* d_out, int out_size, void* d_ws, size_t ws_size,
                              hipStream_t stream) {
  (void)in_sizes; (void)n_in; (void)out_size; (void)ws_size;
  const float* x          = (const float*)d_in[0];
  const float* mask       = (const float*)d_in[1];
  const float* qkv_w      = (const float*)d_in[2];
  const float* qkv_b      = (const float*)d_in[3];
  const float* proj_w     = (const float*)d_in[4];
  const float* proj_b     = (const float*)d_in[5];
  const float* bias_table = (const float*)d_in[6];
  const int*   rel_index  = (const int*)d_in[7];

  float*  cmb     = (float*)((char*)d_ws + CMB_OFF);
  half_t* qkv_wh  = (half_t*)((char*)d_ws + QKVW_OFF);
  half_t* proj_wh = (half_t*)((char*)d_ws + PROJW_OFF);
  half_t* ws_qkv  = (half_t*)((char*)d_ws + WS_QKV_BASE);
  float*  out     = (float*)d_out;

  cvt_w_kernel<<<(576 * 192 + 255) / 256, 256, 0, stream>>>(qkv_w, proj_w,
                                                            qkv_wh, proj_wh);
  cmb_kernel<<<(64 * 6 * 2401 + 255) / 256, 256, 0, stream>>>(mask, bias_table,
                                                              rel_index, cmb);
  qkv_kernel<<<8192, 192, 0, stream>>>(x, qkv_wh, qkv_b, ws_qkv);
  attn_kernel<<<8192, 192, 0, stream>>>(cmb, ws_qkv);
  proj_kernel<<<8192, 192, 0, stream>>>(proj_wh, proj_b, ws_qkv, out);
}
